// MoEHead_24979529793590
// MI455X (gfx1250) — matrix operand order repaired and checked by reading, not compiled, not run on hardware
//
#include <hip/hip_runtime.h>

// ---------------------------------------------------------------------------
// MoE head for MI455X (gfx1250, wave32, WMMA).
//   out[n] = sum_{k<2} softmax(top2(x@Wg.T+bg))_k * (W[e_k] @ x[n] + b[e_k])
// Fused gating + block-local expert grouping + bf16 WMMA with fp32 accum.
// CDNA5 paths: v_wmma_f32_16x16x32_bf16, global_load_async_to_lds_b128 /
// global_store_async_from_lds_b128 (ASYNCcnt), ds_add_f32 LDS atomics.
// ---------------------------------------------------------------------------

typedef __attribute__((ext_vector_type(16))) __bf16 v16bf;
typedef __attribute__((ext_vector_type(8)))  __bf16 v8bf;
typedef __attribute__((ext_vector_type(8)))  float  v8f;

#define DIN       512
#define DOUT      512
#define NEXP      8
#define TILE_N    64          // tokens per block
#define NTHREADS  256         // 8 waves of 32
#define COLS_PER_WAVE 64      // 512 / 8 waves
#define NTILES    4           // 64 cols / 16
#define MAXCH     24          // <= sum ceil(cnt_e/16) <= 16

// Low 32 bits of a generic pointer to __shared__ == LDS byte address
// (flat->LDS address truncation, CDNA5 ISA 10.2).
__device__ __forceinline__ unsigned lds_addr32(const void* p) {
    return (unsigned)(unsigned long long)p;
}

// --- one-time W fp32 -> bf16 conversion into workspace ---------------------
__global__ void moe_wconv_kernel(const float* __restrict__ W,
                                 __bf16* __restrict__ Wb) {
    const int i = (blockIdx.x * blockDim.x + threadIdx.x) * 4;
    const float4 v = *(const float4*)(W + i);
    Wb[i + 0] = (__bf16)v.x;
    Wb[i + 1] = (__bf16)v.y;
    Wb[i + 2] = (__bf16)v.z;
    Wb[i + 3] = (__bf16)v.w;
}

// --- fused MoE kernel -------------------------------------------------------
__global__ __launch_bounds__(NTHREADS, 1)
void moe_head_kernel(const float* __restrict__ x,
                     const __bf16* __restrict__ Wb,   // [E][DOUT][DIN] bf16
                     const float* __restrict__ bvec,  // [E][DOUT]
                     const float* __restrict__ Wg,    // [E][DIN]
                     const float* __restrict__ bg,    // [E]
                     float* __restrict__ out) {
    // LDS: 128KB x-tile + 128KB out accumulator + metadata (~261KB < 320KB)
    __shared__ __align__(16) float xs[TILE_N * DIN];
    __shared__ __align__(16) float outAcc[TILE_N * DOUT];
    __shared__ float gateS[TILE_N][NEXP];
    __shared__ int   tokE[TILE_N][2];
    __shared__ float tokW[TILE_N][2];
    __shared__ int   cntE[NEXP], baseE[NEXP], runE[NEXP];
    __shared__ int   asgTok[TILE_N * 2];
    __shared__ float asgW[TILE_N * 2];
    __shared__ int   chunkE[MAXCH], chunkBase[MAXCH], chunkCnt[MAXCH];
    __shared__ int   nChunks;

    const int tid   = threadIdx.x;
    const int lane  = tid & 31;
    const int wave  = tid >> 5;
    const int tile0 = blockIdx.x * TILE_N;

    // ---- stage x tile into LDS via async copy (ASYNCcnt path) ----
    {
        const unsigned long long xbase =
            (unsigned long long)(x + (size_t)tile0 * DIN);
        const unsigned xsBase = lds_addr32(xs);
#pragma unroll 4
        for (int i = tid; i < TILE_N * DIN / 4; i += NTHREADS) {
            const unsigned voff = (unsigned)i * 16u;   // byte offset per lane
            const unsigned ldsa = xsBase + voff;       // LDS byte dest
            asm volatile("global_load_async_to_lds_b128 %0, %1, %2"
                         :: "v"(ldsa), "v"(voff), "s"(xbase) : "memory");
        }
        asm volatile("s_wait_asynccnt 0x0" ::: "memory");
    }
    if (tid < NEXP) cntE[tid] = 0;
    __syncthreads();

    // ---- gating: wave w handles tokens [8w, 8w+8), lanes split D_IN ----
    for (int tt = 0; tt < 8; ++tt) {
        const int t = wave * 8 + tt;
        float acc[NEXP];
#pragma unroll
        for (int e = 0; e < NEXP; ++e) acc[e] = 0.f;
        for (int d = lane; d < DIN; d += 32) {
            const float xv = xs[t * DIN + d];
#pragma unroll
            for (int e = 0; e < NEXP; ++e)
                acc[e] = fmaf(xv, Wg[e * DIN + d], acc[e]);
        }
#pragma unroll
        for (int e = 0; e < NEXP; ++e) {
            float v = acc[e];
#pragma unroll
            for (int off = 16; off > 0; off >>= 1) v += __shfl_xor(v, off, 32);
            if (lane == 0) gateS[t][e] = v + bg[e];
        }
    }
    __syncthreads();

    // ---- per-token top-2 + softmax (matches jax.lax.top_k tie-breaking) ----
    if (tid < TILE_N) {
        float best = -1e30f, second = -1e30f;
        int bi = 0, si = 0;
#pragma unroll
        for (int e = 0; e < NEXP; ++e) {
            const float s = gateS[tid][e];
            if (s > best)        { second = best; si = bi; best = s; bi = e; }
            else if (s > second) { second = s; si = e; }
        }
        const float e1  = __expf(second - best);
        const float inv = 1.f / (1.f + e1);
        tokE[tid][0] = bi;  tokE[tid][1] = si;
        tokW[tid][0] = inv; tokW[tid][1] = e1 * inv;
        atomicAdd(&cntE[bi], 1);
        atomicAdd(&cntE[si], 1);
    }
    __syncthreads();

    // ---- prefix sums + chunk table (tiny, serial on thread 0) ----
    if (tid == 0) {
        int acc = 0;
        for (int e = 0; e < NEXP; ++e) { baseE[e] = acc; runE[e] = acc; acc += cntE[e]; }
        int nc = 0;
        for (int e = 0; e < NEXP; ++e)
            for (int off = 0; off < cntE[e]; off += 16) {
                chunkE[nc]    = e;
                chunkBase[nc] = baseE[e] + off;
                chunkCnt[nc]  = min(16, cntE[e] - off);
                ++nc;
            }
        nChunks = nc;
    }
    __syncthreads();

    // ---- scatter assignments grouped by expert ----
    if (tid < TILE_N) {
#pragma unroll
        for (int k = 0; k < 2; ++k) {
            const int e   = tokE[tid][k];
            const int pos = atomicAdd(&runE[e], 1);
            asgTok[pos] = tid;
            asgW[pos]   = tokW[tid][k];
        }
    }
    __syncthreads();

    // ---- init output accumulator with weighted biases ----
    for (int idx = tid; idx < TILE_N * DOUT; idx += NTHREADS) {
        const int t = idx >> 9, n = idx & (DOUT - 1);
        outAcc[idx] = tokW[t][0] * bvec[tokE[t][0] * DOUT + n]
                    + tokW[t][1] * bvec[tokE[t][1] * DOUT + n];
    }
    __syncthreads();

    // ---- main loop: bf16 WMMA, each wave owns 64 output columns ----
    const int colBase = wave * COLS_PER_WAVE;
    const int nrow    = lane & 15;       // A: row-in-chunk ; B/D: col-in-16
    const int hi      = (lane >= 16);    // half-wave selector (ISA 7.12.2)
    const int nC      = nChunks;

    for (int c = 0; c < nC; ++c) {
        const int e   = chunkE[c];
        const int cb  = chunkBase[c];
        const int cnt = chunkCnt[c];
        // per-lane gathered A row (pad rows reuse row 0; masked at epilogue)
        const int arow    = (nrow < cnt) ? asgTok[cb + nrow] : asgTok[cb];
        const int arowOff = arow * DIN;

        const __bf16* wbase[NTILES];
#pragma unroll
        for (int nt = 0; nt < NTILES; ++nt)
            wbase[nt] = Wb + ((size_t)e * DOUT + colBase + nt * 16 + nrow) * DIN;

        v8f acc[NTILES] = {};

#pragma unroll 2
        for (int kt = 0; kt < DIN / 32; ++kt) {
            const int kb = kt * 32;
            // A 16x32 bf16: lanes 0-15 hold K {0..7,16..23}, lanes 16-31 {8..15,24..31}
            const int r0 = kb + hi * 8;
            const float4 f0 = *(const float4*)&xs[arowOff + r0];
            const float4 f1 = *(const float4*)&xs[arowOff + r0 + 4];
            const float4 f2 = *(const float4*)&xs[arowOff + r0 + 16];
            const float4 f3 = *(const float4*)&xs[arowOff + r0 + 20];
            v16bf a;
            a[0]  = (__bf16)f0.x; a[1]  = (__bf16)f0.y; a[2]  = (__bf16)f0.z; a[3]  = (__bf16)f0.w;
            a[4]  = (__bf16)f1.x; a[5]  = (__bf16)f1.y; a[6]  = (__bf16)f1.z; a[7]  = (__bf16)f1.w;
            a[8]  = (__bf16)f2.x; a[9]  = (__bf16)f2.y; a[10] = (__bf16)f2.z; a[11] = (__bf16)f2.w;
            a[12] = (__bf16)f3.x; a[13] = (__bf16)f3.y; a[14] = (__bf16)f3.z; a[15] = (__bf16)f3.w;

#pragma unroll
            for (int nt = 0; nt < NTILES; ++nt) {
                // B 32x16 bf16: lane n = lane&15; lanes 0-15 K {0..7,16..23}, 16-31 K {8..15,24..31}
                const v8bf* wp = (const v8bf*)(wbase[nt] + kb + hi * 8);
                const v8bf blo = wp[0];
                const v8bf bhi = wp[2];
                const v16bf bm = __builtin_shufflevector(
                    blo, bhi, 0, 1, 2, 3, 4, 5, 6, 7, 8, 9, 10, 11, 12, 13, 14, 15);
                acc[nt] = __builtin_amdgcn_wmma_f32_16x16x32_bf16(
                    false, a, false, bm, (short)0, acc[nt], false, false);
            }
        }

        // ---- epilogue: weight by softmax gate, accumulate in LDS (ds_add_f32)
#pragma unroll
        for (int r = 0; r < 8; ++r) {
            const int M = r + hi * 8;               // C/D: M = vgpr + 8*(lane>=16)
            if (M < cnt) {
                const int   tok = asgTok[cb + M];
                const float w   = asgW[cb + M];
#pragma unroll
                for (int nt = 0; nt < NTILES; ++nt) {
                    const int n = colBase + nt * 16 + nrow;
                    atomicAdd(&outAcc[tok * DOUT + n], w * acc[nt][r]);
                }
            }
        }
    }
    __syncthreads();

    // ---- write tile to global via async LDS->memory copy ----
    {
        const unsigned long long obase =
            (unsigned long long)(out + (size_t)tile0 * DOUT);
        const unsigned oBase = lds_addr32(outAcc);
#pragma unroll 4
        for (int i = tid; i < TILE_N * DOUT / 4; i += NTHREADS) {
            const unsigned voff = (unsigned)i * 16u;
            const unsigned ldsa = oBase + voff;
            asm volatile("global_store_async_from_lds_b128 %0, %1, %2"
                         :: "v"(voff), "v"(ldsa), "s"(obase) : "memory");
        }
        asm volatile("s_wait_asynccnt 0x0" ::: "memory");
    }
}

// ---------------------------------------------------------------------------
extern "C" void kernel_launch(void* const* d_in, const int* in_sizes, int n_in,
                              void* d_out, int out_size, void* d_ws, size_t ws_size,
                              hipStream_t stream) {
    const float* x  = (const float*)d_in[0];   // [N, DIN]
    const float* W  = (const float*)d_in[1];   // [E, DOUT, DIN]
    const float* b  = (const float*)d_in[2];   // [E, DOUT]
    const float* Wg = (const float*)d_in[3];   // [E, DIN]
    const float* bg = (const float*)d_in[4];   // [E]
    float* out = (float*)d_out;

    const int N = in_sizes[0] / DIN;           // 65536
    __bf16* Wb = (__bf16*)d_ws;                // needs E*DOUT*DIN*2 = 4 MB scratch

    const int wElems = NEXP * DOUT * DIN;      // 2,097,152
    moe_wconv_kernel<<<wElems / 4 / 256, 256, 0, stream>>>(W, Wb);
    moe_head_kernel<<<N / TILE_N, NTHREADS, 0, stream>>>(x, Wb, b, Wg, bg, out);
}
